// TernarizedLinear_31688268709907
// MI455X (gfx1250) — matrix operand order repaired and checked by reading, not compiled, not run on hardware
//
#include <hip/hip_runtime.h>
#include <hip/hip_bf16.h>
#include <stdint.h>

#define N_ROWS 8192
#define K_DIM  4096
#define M_OUT  16384

typedef __attribute__((ext_vector_type(16))) __bf16    v16bf;
typedef __attribute__((ext_vector_type(8)))  float     v8f;
typedef __attribute__((ext_vector_type(8)))  unsigned  v8u;

// pointee/type shapes required by __builtin_amdgcn_global_load_async_to_lds_b128
// (from the round-2 diagnostic: param is "__vector_size__(4*sizeof(int)) int __device__*")
typedef int v4i __attribute__((vector_size(16)));
typedef __attribute__((address_space(1))) v4i* gas_v4i_ptr;
typedef __attribute__((address_space(3))) v4i* las_v4i_ptr;

#if __has_builtin(__builtin_amdgcn_global_load_async_to_lds_b128)
#define HAVE_ASYNC_LDS 1
#else
#define HAVE_ASYNC_LDS 0
#endif

static __device__ __forceinline__ unsigned short f32_to_bf16_rne(float f) {
    unsigned int u = __builtin_bit_cast(unsigned int, f);
    u += 0x7FFFu + ((u >> 16) & 1u);
    return (unsigned short)(u >> 16);
}

// ---------------- threshold = 0.5 * mean(|W|), two deterministic passes ---------------

__global__ void absum_partial(const float4* __restrict__ w, float* __restrict__ partials) {
    __shared__ float red[256];
    const int tid = threadIdx.x;
    const size_t total4 = (size_t)M_OUT * K_DIM / 4;
    size_t idx    = (size_t)blockIdx.x * blockDim.x + tid;
    size_t stride = (size_t)gridDim.x * blockDim.x;
    float s = 0.0f;
    for (size_t i = idx; i < total4; i += stride) {
        float4 v = w[i];
        s += fabsf(v.x) + fabsf(v.y) + fabsf(v.z) + fabsf(v.w);
    }
    red[tid] = s;
    __syncthreads();
    for (int off = 128; off > 0; off >>= 1) {
        if (tid < off) red[tid] += red[tid + off];
        __syncthreads();
    }
    if (tid == 0) partials[blockIdx.x] = red[0];
}

__global__ void absum_final(const float* __restrict__ partials, float* __restrict__ thr_out) {
    __shared__ float red[256];
    const int tid = threadIdx.x;
    float s = 0.0f;
    for (int i = tid; i < 2048; i += 256) s += partials[i];
    red[tid] = s;
    __syncthreads();
    for (int off = 128; off > 0; off >>= 1) {
        if (tid < off) red[tid] += red[tid + off];
        __syncthreads();
    }
    if (tid == 0) thr_out[0] = 0.5f * red[0] / (float)((size_t)M_OUT * K_DIM);
}

// ---------------- ternarize W -> bf16 {-1,0,1}; convert x -> bf16 ---------------------

static __device__ __forceinline__ unsigned short tern_bf16(float f, float thr) {
    unsigned int u = __builtin_bit_cast(unsigned int, f);
    unsigned short sgn = (unsigned short)((u >> 16) & 0x8000u);
    return (fabsf(f) >= thr) ? (unsigned short)(0x3F80u | sgn) : (unsigned short)0;
}

__global__ void ternarize_w(const float4* __restrict__ w, const float* __restrict__ thr_p,
                            uint2* __restrict__ tw) {
    const float thr = thr_p[0];
    size_t i = (size_t)blockIdx.x * blockDim.x + threadIdx.x;
    float4 v = w[i];
    uint2 o;
    o.x = (unsigned)tern_bf16(v.x, thr) | ((unsigned)tern_bf16(v.y, thr) << 16);
    o.y = (unsigned)tern_bf16(v.z, thr) | ((unsigned)tern_bf16(v.w, thr) << 16);
    tw[i] = o;
}

__global__ void convert_x(const float4* __restrict__ x, uint2* __restrict__ xb) {
    size_t i = (size_t)blockIdx.x * blockDim.x + threadIdx.x;
    float4 v = x[i];
    uint2 o;
    o.x = (unsigned)f32_to_bf16_rne(v.x) | ((unsigned)f32_to_bf16_rne(v.y) << 16);
    o.y = (unsigned)f32_to_bf16_rne(v.z) | ((unsigned)f32_to_bf16_rne(v.w) << 16);
    xb[i] = o;
}

// ---------------- bf16 WMMA GEMM: out = Xb @ Wb^T + bias -----------------------------
// Block: 256 threads (8 wave32), 256x128 output tile, K-step 32, double-buffered LDS.
// Wave grid 4(M) x 2(N): each wave -> 64 rows x 64 cols = 4x4 tiles of 16x16
// => 16 v_wmma per K-step against 16 ds_load_b128 (1.0 loads/WMMA).

static __device__ __forceinline__ v8f wmma_bf16(v16bf a, v16bf b, v8f c) {
    return __builtin_amdgcn_wmma_f32_16x16x32_bf16(false, a, false, b, (short)0, c,
                                                   false, false);
}

static __device__ __forceinline__ v16bf pack_frag(uint4 lo, uint4 hi) {
    v8u t;
    t[0] = lo.x; t[1] = lo.y; t[2] = lo.z; t[3] = lo.w;
    t[4] = hi.x; t[5] = hi.y; t[6] = hi.z; t[7] = hi.w;
    return __builtin_bit_cast(v16bf, t);
}

static __device__ __forceinline__ void wait_async0() {
#if HAVE_ASYNC_LDS
#if __has_builtin(__builtin_amdgcn_s_wait_asynccnt)
    __builtin_amdgcn_s_wait_asynccnt(0);
#else
    asm volatile("s_wait_asynccnt 0" ::: "memory");
#endif
#endif
}

#define LDT 40  // padded LDS row stride in bf16 elements (80B -> bank-conflict friendly)
#define BM  256 // block rows (x rows)
#define BN  128 // block cols (weight rows / out cols)

__launch_bounds__(256)
__global__ void gemm_bf16_wmma(const unsigned short* __restrict__ xb,
                               const unsigned short* __restrict__ wb,
                               const float* __restrict__ bias,
                               float* __restrict__ out) {
    __shared__ __align__(16) unsigned short lA[2][BM * LDT];  // 40960 B
    __shared__ __align__(16) unsigned short lB[2][BN * LDT];  // 20480 B

    const int tid  = threadIdx.x;
    const int lane = tid & 31;
    const int wave = tid >> 5;
    const int wm   = wave & 3;   // 0..3 : 64-row strip
    const int wn   = wave >> 2;  // 0..1 : 64-col strip
    const int rowBase = blockIdx.y * BM;
    const int colBase = blockIdx.x * BN;

    const uint4* xg = (const uint4*)xb;  // row stride = 4096/8 = 512 uint4
    const uint4* wg = (const uint4*)wb;

    v8f acc[4][4];
#pragma unroll
    for (int i = 0; i < 4; ++i)
#pragma unroll
        for (int j = 0; j < 4; ++j)
#pragma unroll
            for (int d = 0; d < 8; ++d) acc[i][j][d] = 0.0f;

    // Per stage: A tile 256x32 bf16 = 1024 16B-chunks, B tile 128x32 = 512 chunks.
    // 256 threads -> 4 A-chunks + 2 B-chunks per thread. chunk c: row=c>>2, quad=c&3.

#if HAVE_ASYNC_LDS
    // ---- CDNA5 async global->LDS staging (ASYNCcnt) ----
    auto issue_stage = [&](int buf, int k0) {
        const int kq = k0 >> 3;
#pragma unroll
        for (int i = 0; i < 4; ++i) {
            int c = tid + i * 256;
            int r = c >> 2, q = c & 3;
            __builtin_amdgcn_global_load_async_to_lds_b128(
                (gas_v4i_ptr)(void*)(const_cast<uint4*>(xg) + (size_t)(rowBase + r) * 512 + kq + q),
                (las_v4i_ptr)(void*)&lA[buf][r * LDT + q * 8],
                0, 0);
        }
#pragma unroll
        for (int i = 0; i < 2; ++i) {
            int c = tid + i * 256;
            int r = c >> 2, q = c & 3;
            __builtin_amdgcn_global_load_async_to_lds_b128(
                (gas_v4i_ptr)(void*)(const_cast<uint4*>(wg) + (size_t)(colBase + r) * 512 + kq + q),
                (las_v4i_ptr)(void*)&lB[buf][r * LDT + q * 8],
                0, 0);
        }
    };
    issue_stage(0, 0);
#else
    // ---- fallback: register staging ----
    uint4 ra[4], rb[2];
    auto load_stage_regs = [&](int k0) {
        const int kq = k0 >> 3;
#pragma unroll
        for (int i = 0; i < 4; ++i) {
            int c = tid + i * 256;
            int r = c >> 2, q = c & 3;
            ra[i] = xg[(size_t)(rowBase + r) * 512 + kq + q];
        }
#pragma unroll
        for (int i = 0; i < 2; ++i) {
            int c = tid + i * 256;
            int r = c >> 2, q = c & 3;
            rb[i] = wg[(size_t)(colBase + r) * 512 + kq + q];
        }
    };
    load_stage_regs(0);
#endif

    int buf = 0;
    for (int k0 = 0; k0 < K_DIM; k0 += 32) {
#if HAVE_ASYNC_LDS
        wait_async0();            // own async copies for stage k0 done
        __syncthreads();          // everyone's copies visible
        if (k0 + 32 < K_DIM) issue_stage(buf ^ 1, k0 + 32);
#else
        // stage registers -> LDS  (element off = r*LDT + q*8 -> uint4 idx = r*5 + q)
        uint4* la = (uint4*)&lA[buf][0];
        uint4* lb = (uint4*)&lB[buf][0];
#pragma unroll
        for (int i = 0; i < 4; ++i) {
            int c = tid + i * 256;
            la[(c >> 2) * 5 + (c & 3)] = ra[i];
        }
#pragma unroll
        for (int i = 0; i < 2; ++i) {
            int c = tid + i * 256;
            lb[(c >> 2) * 5 + (c & 3)] = rb[i];
        }
        __syncthreads();
        if (k0 + 32 < K_DIM) load_stage_regs(k0 + 32);
#endif

        // A fragments (ISA 16-bit A 16x32 layout: lanes 0-15 K[0..7],K[16..23];
        // lanes 16-31 K[8..15],K[24..31])
        const uint4* car = (const uint4*)&lA[buf][0];
        const uint4* cbr = (const uint4*)&lB[buf][0];
        const int sel = lane >> 4;
        v16bf afr[4];
#pragma unroll
        for (int tm = 0; tm < 4; ++tm) {
            int r = wm * 64 + tm * 16 + (lane & 15);
            uint4 lo = car[r * 5 + sel];       // K chunk sel*8 .. sel*8+7
            uint4 hi = car[r * 5 + sel + 2];   // +16 elements
            afr[tm] = pack_frag(lo, hi);
        }
        // B fragments (lane = N column, lanes 0-15: K[0..7],K[16..23]; lanes 16-31: K[8..15],K[24..31])
        v16bf bfr[4];
#pragma unroll
        for (int tn = 0; tn < 4; ++tn) {
            int n = wn * 64 + tn * 16 + (lane & 15);
            uint4 lo = cbr[n * 5 + sel];
            uint4 hi = cbr[n * 5 + sel + 2];
            bfr[tn] = pack_frag(lo, hi);
        }

#pragma unroll
        for (int tm = 0; tm < 4; ++tm)
#pragma unroll
            for (int tn = 0; tn < 4; ++tn)
                acc[tm][tn] = wmma_bf16(afr[tm], bfr[tn], acc[tm][tn]);

        buf ^= 1;
    }

    // epilogue: C/D layout -> VGPR d: M = d + 8*(lane>>4), N = lane&15 ; add bias
#pragma unroll
    for (int tn = 0; tn < 4; ++tn) {
        const int col = colBase + wn * 64 + tn * 16 + (lane & 15);
        const float bv = bias[col];
        const int rbase = rowBase + wm * 64 + (lane >> 4) * 8;
#pragma unroll
        for (int tm = 0; tm < 4; ++tm) {
#pragma unroll
            for (int d = 0; d < 8; ++d) {
                int row = rbase + tm * 16 + d;
                out[(size_t)row * M_OUT + col] = acc[tm][tn][d] + bv;
            }
        }
    }
}

// ---------------------------------- launch -------------------------------------------

extern "C" void kernel_launch(void* const* d_in, const int* in_sizes, int n_in,
                              void* d_out, int out_size, void* d_ws, size_t ws_size,
                              hipStream_t stream) {
    const float* x    = (const float*)d_in[0];
    const float* w    = (const float*)d_in[1];
    const float* bias = (const float*)d_in[2];
    float* out        = (float*)d_out;

    char* ws = (char*)d_ws;
    float* partials       = (float*)ws;               // 2048 floats
    float* thr            = (float*)(ws + 8192);      // 1 float
    unsigned short* xb    = (unsigned short*)(ws + 16384);                    // 64 MiB
    unsigned short* wbuf  = xb + (size_t)N_ROWS * K_DIM;                      // 128 MiB

    absum_partial<<<2048, 256, 0, stream>>>((const float4*)w, partials);
    absum_final<<<1, 256, 0, stream>>>(partials, thr);

    ternarize_w<<<((size_t)M_OUT * K_DIM / 4) / 256, 256, 0, stream>>>(
        (const float4*)w, thr, (uint2*)wbuf);
    convert_x<<<((size_t)N_ROWS * K_DIM / 4) / 256, 256, 0, stream>>>(
        (const float4*)x, (uint2*)xb);

    dim3 grid(M_OUT / BN, N_ROWS / BM);
    gemm_bf16_wmma<<<grid, 256, 0, stream>>>(xb, wbuf, bias, out);
}
